// FBPINN_27977416966167
// MI455X (gfx1250) — matrix operand order repaired and checked by reading, not compiled, not run on hardware
//
#include <hip/hip_runtime.h>
#include <hip/hip_fp16.h>

// FBPINN on MI455X (gfx1250): batched (32768x128)@(128x128) GEMMs on the WMMA
// pipe (f16 in / f32 acc), weights staged to LDS by the Tensor Data Mover with
// hardware row padding, tanh on the v_tanh_f32 transcendental pipe.

typedef __attribute__((ext_vector_type(16))) _Float16 v16h;
typedef __attribute__((ext_vector_type(8)))  _Float16 v8h;
typedef __attribute__((ext_vector_type(8)))  float    v8f;
typedef __attribute__((ext_vector_type(4)))  float    v4f;
typedef __attribute__((ext_vector_type(4)))  unsigned v4u;
typedef __attribute__((ext_vector_type(8)))  unsigned v8u;

#define NSUB   16
#define H      128
#define N_MID  3
#define N_PTS  32768
#define TILE_M 128
#define ROWH   136   // padded LDS row stride in halves (256B data + 16B pad)
#define SIGMA  0.005f
#define WIDTH  (1.0f/16.0f)
#define W_FREQ 15.0f

__device__ __forceinline__ float fast_tanh(float v) {
#if __has_builtin(__builtin_amdgcn_tanhf)
  return __builtin_amdgcn_tanhf(v);          // v_tanh_f32 (TRANS pipe, co-executes with WMMA)
#elif __has_builtin(__builtin_amdgcn_tanh_f32)
  return __builtin_amdgcn_tanh_f32(v);
#else
  float e = __expf(2.0f * v);
  return 1.0f - 2.0f / (e + 1.0f);
#endif
}

// ---- TDM: load one 128x128 f16 weight tile into LDS with 16B pad per 256B row ----
// D# group0/group1 per CDNA5 ISA 8.3/8.4; 2D tensor -> groups 2/3 omitted.
__device__ __forceinline__ void tdm_load_weights(const _Float16* gsrc, unsigned lds_off) {
  unsigned long long ga = (unsigned long long)(uintptr_t)gsrc;
  v4u g0;
  g0[0] = 1u;                                   // count=1 (valid), user mode
  g0[1] = lds_off;                              // lds_addr
  g0[2] = (unsigned)ga;                         // global_addr[31:0]
  g0[3] = (unsigned)(ga >> 32) | (2u << 30);    // global_addr[56:32] | type=2 (image)
  v8u g1;
  g1[0] = (1u << 16)                            // data_size = 1 -> 2-byte elements
        | (1u << 20)                            // pad_enable
        | (5u << 22)                            // pad_interval: 2^5 * 8B = 256B
        | (3u << 25);                           // pad_amount: 4 DWORDs = 16B
  g1[1] = (128u << 16);                         // tensor_dim0[15:0] (bits 63:48)
  g1[2] = (128u << 16);                         // tensor_dim1[15:0] (bits 95:80)
  g1[3] = (128u << 16);                         // tile_dim0 = 128   (bits 127:112)
  g1[4] = 128u;                                 // tile_dim1 = 128   (bits 143:128)
  g1[5] = 128u;                                 // tensor_dim0_stride = 128 elems
  g1[6] = 0u;
  g1[7] = 0u;
  asm volatile("tensor_load_to_lds %0, %1" :: "s"(g0), "s"(g1) : "memory");
}

// ---------------- pass 1: min/max of x ----------------
__global__ void minmax_kernel(const float* __restrict__ x, float* __restrict__ mm) {
  __shared__ float smin[1024];
  __shared__ float smax[1024];
  int t = threadIdx.x;
  float mn = 1e30f, mx = -1e30f;
  for (int i = t; i < N_PTS; i += 1024) {
    float v = x[i];
    mn = fminf(mn, v);
    mx = fmaxf(mx, v);
  }
  smin[t] = mn; smax[t] = mx;
  __syncthreads();
  for (int s = 512; s > 0; s >>= 1) {
    if (t < s) {
      smin[t] = fminf(smin[t], smin[t + s]);
      smax[t] = fmaxf(smax[t], smax[t + s]);
    }
    __syncthreads();
  }
  if (t == 0) { mm[0] = smin[0]; mm[1] = smax[0]; }
}

// ---------------- pass 2: W_h fp32 -> fp16, transposed to Wt[s][l][out][in] ----------------
__global__ void convert_kernel(const float* __restrict__ Wh, _Float16* __restrict__ Wt) {
  int i = blockIdx.x * 256 + threadIdx.x;            // dst linear index
  if (i >= NSUB * N_MID * H * H) return;
  int in  = i & 127;
  int out = (i >> 7) & 127;
  int sl  = i >> 14;
  Wt[i] = (_Float16)Wh[(sl * H + in) * H + out];     // src is [sl][in][out]
}

// ---------------- main kernel ----------------
__global__ __launch_bounds__(256) void fbpinn_kernel(
    const float* __restrict__ x,
    const float* __restrict__ W_in,
    const float* __restrict__ b_in,
    const _Float16* __restrict__ Wt,     // [s][l][out][in] f16
    const float* __restrict__ b_h,
    const float* __restrict__ W_out,
    const float* __restrict__ b_out,
    const float* __restrict__ mm,
    float* __restrict__ out)
{
  extern __shared__ char smem[];
  _Float16* hA  = (_Float16*)smem;                // 128 x ROWH halves (34816 B)
  _Float16* hB  = hA  + TILE_M * ROWH;
  _Float16* Wl0 = hB  + TILE_M * ROWH;            // weight tile, double-buffered
  _Float16* Wl1 = Wl0 + H * ROWH;
  float*    xs  = (float*)(Wl1 + H * ROWH);       // 128 raw x values

  const int tid  = threadIdx.x;
  const int lane = tid & 31;
  const int wave = tid >> 5;
  const int p0   = blockIdx.x * TILE_M;

  const float xmin = mm[0];
  const float xmax = mm[1];
  const float inv2 = 2.0f / (xmax - xmin);

  const unsigned wloff0 = (unsigned)(uintptr_t)(void*)Wl0;
  const unsigned wloff1 = (unsigned)(uintptr_t)(void*)Wl1;

  if (tid < TILE_M) xs[tid] = x[p0 + tid];

  // kick off TDM for the first weight tile (wave 0 only; uniform descriptors)
  if (wave == 0) tdm_load_weights(Wt, wloff0);
  __syncthreads();

  // output accumulator: even threads own point (tid>>1)
  const int myp = tid >> 1;
  const int hv  = tid & 1;
  float outacc = 0.0f;

  // per-wave fragment geometry (16-bit A layout / B layout per CDNA5 ISA 7.12.2)
  const int mbase = wave * 16;
  const int arow  = mbase + (lane & 15);
  const int akoff = (lane & 16) ? 8 : 0;    // A: lanes 16-31 take K +8 within each 16-chunk
  const int bkoff = (lane & 16) ? 8 : 0;    // B: lanes 16-31 take K +8 within each 16-chunk
  const int ncol  = lane & 15;
  const int mofs  = (lane & 16) ? 8 : 0;    // C/D: lanes 16-31 hold rows M+8

  for (int s = 0; s < NSUB; ++s) {
    // ---- layer 0: h = tanh(xn * W_in + b_in) -> hA (f16) ----
    {
      int n = tid & 127;
      float wi = W_in[s * H + n];
      float bi = b_in[s * H + n];
      for (int m = (tid >> 7); m < TILE_M; m += 2) {
        float xn = (xs[m] - xmin) * inv2 - 1.0f;
        hA[m * ROWH + n] = (_Float16)fast_tanh(xn * wi + bi);
      }
    }

    _Float16* cur = hA;
    _Float16* nxt = hB;

    for (int l = 0; l < N_MID; ++l) {
      const int it = s * N_MID + l;
      if (wave == 0) __builtin_amdgcn_s_wait_tensorcnt(0);  // weight tile `it` landed
      __syncthreads();   // weights visible to all waves; prev GEMM fully retired
      if (wave == 0 && (it + 1) < NSUB * N_MID) {
        // prefetch next layer's weights into the other buffer (safe: last read
        // of that buffer was GEMM it-1, which all threads finished above)
        tdm_load_weights(Wt + (size_t)(it + 1) * H * H,
                         ((it + 1) & 1) ? wloff1 : wloff0);
      }
      const _Float16* Wl = (it & 1) ? Wl1 : Wl0;

      // ---- GEMM: wave computes rows [mbase, mbase+16) x all 128 cols ----
      v8f acc[8] = {};
      const float* bptr = b_h + (size_t)(s * N_MID + l) * H;

      #pragma unroll
      for (int kt = 0; kt < 4; ++kt) {
        union { v16h v; v8h h8[2]; } afrag;
        const _Float16* ap = cur + arow * ROWH + kt * 32 + akoff;
        afrag.h8[0] = *(const v8h*)(ap);        // K = base+0..7   (VGPR0-3)
        afrag.h8[1] = *(const v8h*)(ap + 16);   // K = base+16..23 (VGPR4-7)
        #pragma unroll
        for (int t = 0; t < 8; ++t) {
          union { v16h v; v8h h8[2]; } bfrag;
          const _Float16* bp = Wl + (t * 16 + ncol) * ROWH + kt * 32 + bkoff;
          bfrag.h8[0] = *(const v8h*)(bp);
          bfrag.h8[1] = *(const v8h*)(bp + 16);
          acc[t] = __builtin_amdgcn_wmma_f32_16x16x32_f16(
              false, afrag.v, false, bfrag.v, (short)0, acc[t], false, false);
        }
      }

      // ---- epilogue: bias + tanh, write f16 to nxt ----
      #pragma unroll
      for (int t = 0; t < 8; ++t) {
        int n = t * 16 + ncol;
        float bias = bptr[n];
        #pragma unroll
        for (int r = 0; r < 8; ++r) {
          float v = fast_tanh(acc[t][r] + bias);
          nxt[(mbase + mofs + r) * ROWH + n] = (_Float16)v;
        }
      }

      _Float16* tmp = cur; cur = nxt; nxt = tmp;
    }

    __syncthreads();   // final activations (in cur == hB) visible

    // ---- output layer + window blend (vectorized dot) ----
    {
      const float* wo = W_out + s * H;
      float sum = 0.0f;
      int jb = hv * 64;
      #pragma unroll
      for (int c = 0; c < 8; ++c) {
        v8h hvv = *(const v8h*)(cur + myp * ROWH + jb + c * 8);
        v4f w0 = *(const v4f*)(wo + jb + c * 8);
        v4f w1 = *(const v4f*)(wo + jb + c * 8 + 4);
        sum += (float)hvv[0] * w0[0] + (float)hvv[1] * w0[1]
             + (float)hvv[2] * w0[2] + (float)hvv[3] * w0[3]
             + (float)hvv[4] * w1[0] + (float)hvv[5] * w1[1]
             + (float)hvv[6] * w1[2] + (float)hvv[7] * w1[3];
      }
      sum += __shfl_xor(sum, 1, 32);   // pair (2p, 2p+1) within a wave32
      if (hv == 0) {
        float u  = sum + b_out[s];
        float xv = xs[myp];
        float wa = (float)s * WIDTH;
        float wb = (float)(s + 1) * WIDTH;
        float left  = 1.0f / (1.0f + __expf(-(xv - wa) / SIGMA));
        float right = 1.0f / (1.0f + __expf(-(wb - xv) / SIGMA));
        float win = (s == 0) ? right : ((s == NSUB - 1) ? left : left * right);
        outacc += win * u;
      }
    }
  }

  if (hv == 0) out[p0 + myp] = outacc * (1.0f / W_FREQ);
}

// ---------------- launcher ----------------
extern "C" void kernel_launch(void* const* d_in, const int* in_sizes, int n_in,
                              void* d_out, int out_size, void* d_ws, size_t ws_size,
                              hipStream_t stream) {
  const float* x     = (const float*)d_in[0];
  const float* W_in  = (const float*)d_in[1];
  const float* b_in  = (const float*)d_in[2];
  const float* W_h   = (const float*)d_in[3];
  const float* b_h   = (const float*)d_in[4];
  const float* W_out = (const float*)d_in[5];
  const float* b_out = (const float*)d_in[6];
  float* out = (float*)d_out;

  float*     mm = (float*)d_ws;                          // [2]: xmin, xmax
  _Float16*  Wt = (_Float16*)((char*)d_ws + 256);        // 16*3*128*128 f16 (1.5MB)

  minmax_kernel<<<1, 1024, 0, stream>>>(x, mm);

  int nW = NSUB * N_MID * H * H;
  convert_kernel<<<nW / 256, 256, 0, stream>>>(W_h, Wt);

  size_t lds = (size_t)(2 * TILE_M * ROWH + 2 * H * ROWH) * sizeof(_Float16)
             + (size_t)TILE_M * sizeof(float);           // ~137 KB of the 320 KB WGP LDS
  fbpinn_kernel<<<N_PTS / TILE_M, 256, lds, stream>>>(
      x, W_in, b_in, Wt, b_h, W_out, b_out, mm, out);
}
